// DecoderRNN_15925738734170
// MI455X (gfx1250) — matrix operand order repaired and checked by reading, not compiled, not run on hardware
//
#include <hip/hip_runtime.h>
#include <hip/hip_bf16.h>
#include <math.h>

// ---------------------------------------------------------------------------
// Sizes (fixed by the reference problem)
// ---------------------------------------------------------------------------
#define H_  1024
#define V_  32000
#define B_  16
#define T_  256
#define K3_ 3072           // 3*H
#define M_  (T_*B_)        // 4096 rows (t-major, batch-minor)
#define SOS_ 1

typedef __attribute__((ext_vector_type(16))) __bf16 v16bf;
typedef __attribute__((ext_vector_type(8)))  float  v8f;

union ABFrag { uint4 u[2]; v16bf v; };

// A-fragment (16x32, M x K), wave32 layout:
//   lane L: m = L%16, g = L/16; vgprs 0..3 = K[kb+g*8 .. +8), vgprs 4..7 = K[kb+16+g*8 .. +8)
__device__ __forceinline__ v16bf load_a_frag(const __bf16* __restrict__ Arow, int kb, int g) {
    ABFrag f;
    f.u[0] = *(const uint4*)(Arow + kb + g * 8);
    f.u[1] = *(const uint4*)(Arow + kb + 16 + g * 8);
    return f.v;
}

// B-fragment (32x16, K x N) from row-major W[N][K] (B[k][n] = W[n][k]):
//   lane L: n = L%16, g = L/16; vgprs 0..3 = K[kb+g*8 .. +8), vgprs 4..7 = K[kb+16+g*8 .. +8)
__device__ __forceinline__ v16bf load_b_frag(const __bf16* __restrict__ Wrow, int kb, int g) {
    ABFrag f;
    f.u[0] = *(const uint4*)(Wrow + kb + g * 8);
    f.u[1] = *(const uint4*)(Wrow + kb + 16 + g * 8);
    return f.v;
}

__device__ __forceinline__ v8f wmma_bf16(v16bf a, v16bf b, v8f c) {
    return __builtin_amdgcn_wmma_f32_16x16x32_bf16(false, a, false, b, (short)0, c, false, false);
}

// ---------------------------------------------------------------------------
// fp32 -> bf16 convert (weights, one-time)
// ---------------------------------------------------------------------------
__global__ void f32_to_bf16_kernel(const float* __restrict__ src, __bf16* __restrict__ dst, size_t n) {
    size_t i = (size_t)blockIdx.x * blockDim.x + threadIdx.x;
    if (i < n) dst[i] = (__bf16)src[i];
}

// ---------------------------------------------------------------------------
// Embedding + ReLU -> bf16 activations X[t*B+b][k]
// ---------------------------------------------------------------------------
__global__ void embed_relu_kernel(const int* __restrict__ target, const float* __restrict__ emb,
                                  __bf16* __restrict__ Xbf) {
    size_t idx = (size_t)blockIdx.x * blockDim.x + threadIdx.x;
    if (idx >= (size_t)M_ * H_) return;
    int k   = (int)(idx % H_);
    size_t row = idx / H_;
    int b   = (int)(row % B_);
    int t   = (int)(row / B_);
    int tok = (t == 0) ? SOS_ : target[b * T_ + (t - 1)];
    float v = emb[(size_t)tok * H_ + k];
    v = v > 0.f ? v : 0.f;
    Xbf[idx] = (__bf16)v;
}

// h0 init (fp32 + bf16 copy into buffer 0)
__global__ void init_h_kernel(const float* __restrict__ enc_hid, float* __restrict__ hf,
                              __bf16* __restrict__ hbf0) {
    int i = blockIdx.x * blockDim.x + threadIdx.x;
    if (i >= B_ * H_) return;
    float v = enc_hid[i];
    hf[i]   = v;
    hbf0[i] = (__bf16)v;
}

// ---------------------------------------------------------------------------
// Generic bf16 GEMM (K = 1024 compile-time):
//   C[m][n] = sum_k A[m][k]*W[n][k] + bias[n]
//   block = 128 threads (4 waves); wave computes 32(M) x 64(N): 8 accumulators,
//   2 A-frags x 4 B-frags per K-slice. Ping-pong double buffering with
//   IN-PLACE reloads (no register rotation copies -> no mov/NOP chains):
//     compute(buf0); reload buf0 <- slice+2; compute(buf1); reload buf1 <- slice+3
//   grid = (N/256, M/32).  TRANS!=0 remaps output row m=t*B+b -> b*T+t.
// ---------------------------------------------------------------------------
#define LOAD_SLICE(A0, A1, Bv0, Bv1, Bv2, Bv3, KB)  \
    do {                                            \
        A0  = load_a_frag(Arow0, (KB), g);          \
        A1  = load_a_frag(Arow1, (KB), g);          \
        Bv0 = load_b_frag(Wr0,   (KB), g);          \
        Bv1 = load_b_frag(Wr1,   (KB), g);          \
        Bv2 = load_b_frag(Wr2,   (KB), g);          \
        Bv3 = load_b_frag(Wr3,   (KB), g);          \
    } while (0)

#define COMPUTE_SLICE(A0, A1, Bv0, Bv1, Bv2, Bv3)   \
    do {                                            \
        acc0 = wmma_bf16(A0, Bv0, acc0);            \
        acc4 = wmma_bf16(A1, Bv0, acc4);            \
        acc1 = wmma_bf16(A0, Bv1, acc1);            \
        acc5 = wmma_bf16(A1, Bv1, acc5);            \
        acc2 = wmma_bf16(A0, Bv2, acc2);            \
        acc6 = wmma_bf16(A1, Bv2, acc6);            \
        acc3 = wmma_bf16(A0, Bv3, acc3);            \
        acc7 = wmma_bf16(A1, Bv3, acc7);            \
    } while (0)

template <int TRANS>
__global__ __launch_bounds__(128)
void gemm_bf16_nt(const __bf16* __restrict__ A, const __bf16* __restrict__ W,
                  const float* __restrict__ bias, float* __restrict__ C, int ldc) {
    constexpr int K = H_;   // 1024 in every matmul of this model
    int wave = threadIdx.x >> 5;
    int lane = threadIdx.x & 31;
    int g    = lane >> 4;     // 0/1 (lane half)
    int ml   = lane & 15;
    int m_base = blockIdx.y * 32;
    int n_base = blockIdx.x * 256 + wave * 64;

    const __bf16* Arow0 = A + (size_t)(m_base + ml) * K;
    const __bf16* Arow1 = Arow0 + (size_t)16 * K;
    const __bf16* Wr0   = W + (size_t)(n_base + ml) * K;
    const __bf16* Wr1   = Wr0 + (size_t)16 * K;
    const __bf16* Wr2   = Wr0 + (size_t)32 * K;
    const __bf16* Wr3   = Wr0 + (size_t)48 * K;

    v8f acc0 = {}, acc1 = {}, acc2 = {}, acc3 = {};
    v8f acc4 = {}, acc5 = {}, acc6 = {}, acc7 = {};

    v16bf a0, a1, b0, b1, b2, b3;        // buffer 0
    v16bf c0, c1, d0, d1, d2, d3;        // buffer 1

    LOAD_SLICE(a0, a1, b0, b1, b2, b3, 0);
    LOAD_SLICE(c0, c1, d0, d1, d2, d3, 32);

    for (int kb = 0; kb < K - 64; kb += 64) {
        COMPUTE_SLICE(a0, a1, b0, b1, b2, b3);
        LOAD_SLICE(a0, a1, b0, b1, b2, b3, kb + 64);
        COMPUTE_SLICE(c0, c1, d0, d1, d2, d3);
        LOAD_SLICE(c0, c1, d0, d1, d2, d3, kb + 96);
    }
    COMPUTE_SLICE(a0, a1, b0, b1, b2, b3);
    COMPUTE_SLICE(c0, c1, d0, d1, d2, d3);

    v8f accs[8] = {acc0, acc1, acc2, acc3, acc4, acc5, acc6, acc7};
#pragma unroll
    for (int mt = 0; mt < 2; ++mt) {
#pragma unroll
        for (int tN = 0; tN < 4; ++tN) {
            int n = n_base + tN * 16 + ml;
            float bv = bias[n];
#pragma unroll
            for (int r = 0; r < 8; ++r) {
                int m = m_base + mt * 16 + g * 8 + r;
                int row = TRANS ? ((m & (B_ - 1)) * T_ + (m >> 4)) : m;
                C[(size_t)row * ldc + n] = accs[mt * 4 + tN][r] + bv;
            }
        }
    }
}

// ---------------------------------------------------------------------------
// One GRU step: gh = h @ W_hh^T + b_hh, then fused gate math.
//   64 blocks x 128 threads; block owns 16 h-columns c..c+16 and computes the
//   three matching gate tiles (r at c, z at c+H, n at c+2H) with waves 0..2.
//   K-loop fully unrolled (32 slices, constant offsets). h double-buffered.
// ---------------------------------------------------------------------------
__global__ __launch_bounds__(128)
void gru_step_kernel(int t,
                     const __bf16* __restrict__ hbf_in, __bf16* __restrict__ hbf_out,
                     float* __restrict__ hf,
                     const __bf16* __restrict__ Whh, const float* __restrict__ bhh,
                     const float* __restrict__ Gi, __bf16* __restrict__ Hs) {
    __shared__ float gh[3][16][16];   // [gate][batch][col]

    int c    = blockIdx.x * 16;
    int wave = threadIdx.x >> 5;
    int lane = threadIdx.x & 31;
    int g    = lane >> 4;
    int ml   = lane & 15;

    if (wave < 3) {
        v8f acc = {};
        const __bf16* Arow = hbf_in + (size_t)ml * H_;   // h row = batch ml
        const __bf16* Wb   = Whh + (size_t)(wave * H_ + c + ml) * H_;
#pragma unroll
        for (int kb = 0; kb < H_; kb += 32) {
            v16bf a = load_a_frag(Arow, kb, g);
            v16bf b = load_b_frag(Wb, kb, g);
            acc = wmma_bf16(a, b, acc);
        }
        float bv = bhh[wave * H_ + c + ml];
#pragma unroll
        for (int r = 0; r < 8; ++r)
            gh[wave][g * 8 + r][ml] = acc[r] + bv;
    }
    __syncthreads();

    // gate math: 16x16 = 256 elements, 2 per thread
    for (int e = threadIdx.x; e < 256; e += 128) {
        int b  = e >> 4;
        int jl = e & 15;
        int j  = c + jl;
        size_t gib = (size_t)(t * B_ + b) * K3_ + j;
        float ir  = Gi[gib];
        float iz  = Gi[gib + H_];
        float in_ = Gi[gib + 2 * H_];
        float hr = gh[0][b][jl];
        float hz = gh[1][b][jl];
        float hn = gh[2][b][jl];
        float r = 1.f / (1.f + __expf(-(ir + hr)));
        float z = 1.f / (1.f + __expf(-(iz + hz)));
        float n = tanhf(in_ + r * hn);
        float ho = hf[b * H_ + j];
        float hnew = (1.f - z) * n + z * ho;
        hf[b * H_ + j] = hnew;
        __bf16 hb = (__bf16)hnew;
        hbf_out[b * H_ + j] = hb;
        Hs[(size_t)(t * B_ + b) * H_ + j] = hb;
    }
}

// ---------------------------------------------------------------------------
// In-place log-softmax over V per row; one 256-thread block per (b,t) row.
// ---------------------------------------------------------------------------
__global__ __launch_bounds__(256)
void logsoftmax_kernel(float* __restrict__ out) {
    __shared__ float red[256];
    float* p = out + (size_t)blockIdx.x * V_;
    int tid = threadIdx.x;

    float mx = -INFINITY;
    for (int i = tid; i < V_; i += 256) mx = fmaxf(mx, p[i]);
    red[tid] = mx;
    __syncthreads();
    for (int s = 128; s > 0; s >>= 1) {
        if (tid < s) red[tid] = fmaxf(red[tid], red[tid + s]);
        __syncthreads();
    }
    mx = red[0];
    __syncthreads();

    float sum = 0.f;
    for (int i = tid; i < V_; i += 256) sum += __expf(p[i] - mx);
    red[tid] = sum;
    __syncthreads();
    for (int s = 128; s > 0; s >>= 1) {
        if (tid < s) red[tid] += red[tid + s];
        __syncthreads();
    }
    float lse = __logf(red[0]) + mx;
    __syncthreads();

    for (int i = tid; i < V_; i += 256) p[i] = p[i] - lse;
}

__global__ void write_hfinal_kernel(const float* __restrict__ hf, float* __restrict__ out) {
    int i = blockIdx.x * blockDim.x + threadIdx.x;
    if (i < B_ * H_) out[(size_t)B_ * T_ * V_ + i] = hf[i];
}

// ---------------------------------------------------------------------------
// Host launcher
// ---------------------------------------------------------------------------
extern "C" void kernel_launch(void* const* d_in, const int* in_sizes, int n_in,
                              void* d_out, int out_size, void* d_ws, size_t ws_size,
                              hipStream_t stream) {
    (void)in_sizes; (void)n_in; (void)out_size; (void)ws_size;
    const float* enc_hid = (const float*)d_in[1];
    const int*   target  = (const int*)d_in[2];
    const float* emb     = (const float*)d_in[3];
    const float* W_ih    = (const float*)d_in[4];
    const float* W_hh    = (const float*)d_in[5];
    const float* b_ih    = (const float*)d_in[6];
    const float* b_hh    = (const float*)d_in[7];
    const float* W_out   = (const float*)d_in[8];
    const float* b_out   = (const float*)d_in[9];
    float* out = (float*)d_out;

    // workspace carve-up (~146 MB)
    char* ws = (char*)d_ws;
    size_t off = 0;
    auto carve = [&](size_t bytes) -> void* {
        void* p = ws + off;
        off = (off + bytes + 255) & ~(size_t)255;
        return p;
    };
    __bf16* Xbf    = (__bf16*)carve((size_t)M_ * H_ * 2);       // 8 MB
    __bf16* Wih_b  = (__bf16*)carve((size_t)K3_ * H_ * 2);      // 6 MB
    __bf16* Whh_b  = (__bf16*)carve((size_t)K3_ * H_ * 2);      // 6 MB
    __bf16* Wout_b = (__bf16*)carve((size_t)V_ * H_ * 2);       // 64 MB (L2-resident)
    float*  Gi     = (float*)carve((size_t)M_ * K3_ * 4);       // 50 MB
    __bf16* Hs     = (__bf16*)carve((size_t)M_ * H_ * 2);       // 8 MB
    float*  hf     = (float*)carve((size_t)B_ * H_ * 4);
    __bf16* hbf0   = (__bf16*)carve((size_t)B_ * H_ * 2);       // double buffer
    __bf16* hbf1   = (__bf16*)carve((size_t)B_ * H_ * 2);

    // 1) one-time bf16 weight conversion
    {
        size_t n1 = (size_t)K3_ * H_;
        size_t n3 = (size_t)V_ * H_;
        f32_to_bf16_kernel<<<(unsigned)((n1 + 255) / 256), 256, 0, stream>>>(W_ih, Wih_b, n1);
        f32_to_bf16_kernel<<<(unsigned)((n1 + 255) / 256), 256, 0, stream>>>(W_hh, Whh_b, n1);
        f32_to_bf16_kernel<<<(unsigned)((n3 + 255) / 256), 256, 0, stream>>>(W_out, Wout_b, n3);
    }

    // 2) embedding + relu -> bf16 activations; h0 init
    {
        size_t n = (size_t)M_ * H_;
        embed_relu_kernel<<<(unsigned)((n + 255) / 256), 256, 0, stream>>>(target, emb, Xbf);
        init_h_kernel<<<(B_ * H_ + 255) / 256, 256, 0, stream>>>(enc_hid, hf, hbf0);
    }

    // 3) Gi = X @ W_ih^T + b_ih  (4096 x 3072, K=1024) — hoisted out of recurrence
    {
        dim3 grid(K3_ / 256, M_ / 32);
        gemm_bf16_nt<0><<<grid, 128, 0, stream>>>(Xbf, Wih_b, b_ih, Gi, K3_);
    }

    // 4) sequential GRU recurrence: 256 small fused launches, h double-buffered
    for (int t = 0; t < T_; ++t) {
        const __bf16* hin  = (t & 1) ? hbf1 : hbf0;
        __bf16*       hout = (t & 1) ? hbf0 : hbf1;
        gru_step_kernel<<<K3_ / 16 / 3, 128, 0, stream>>>(t, hin, hout, hf,
                                                          Whh_b, b_hh, Gi, Hs);
    }

    // 5) logits = Hs @ W_out^T + b_out, written transposed into d_out as [B,T,V]
    {
        dim3 grid(V_ / 256, M_ / 32);
        gemm_bf16_nt<1><<<grid, 128, 0, stream>>>(Hs, Wout_b, b_out, out, V_);
    }

    // 6) in-place log-softmax per (b,t) row; append h_final
    logsoftmax_kernel<<<B_ * T_, 256, 0, stream>>>(out);
    write_hfinal_kernel<<<(B_ * H_ + 255) / 256, 256, 0, stream>>>(hf, out);
}
